// sentence_encoder_23974507446844
// MI455X (gfx1250) — compile-verified, matrix operand order checked by reading, not run on hardware
//
#include <hip/hip_runtime.h>
#include <hip/hip_bf16.h>

// ---- types -----------------------------------------------------------------
typedef __attribute__((ext_vector_type(16))) __bf16       v16bf;
typedef __attribute__((ext_vector_type(8)))  float        v8f;
typedef __attribute__((ext_vector_type(4)))  float        v4f;
typedef __attribute__((ext_vector_type(4)))  unsigned int v4u;
typedef __attribute__((ext_vector_type(4)))  __bf16       v4bf;

namespace {

constexpr int kH      = 1024;   // hidden/embed dim
constexpr int kG      = 4096;   // 4*H (i,f,g,o)
constexpr int kL      = 8192;   // tokens
constexpr int kLayers = 2;
constexpr int kKS     = 32;     // K-step per WMMA round
constexpr int kLdsRS  = 40;     // padded LDS row stride (bf16 elems, 80B = 20 banks)

union FragU { v16bf v; v4u q[2]; };

// Per-lane WMMA A/B fragment load for row-major [row][K] bf16 data.
// lane -> row = lane&15; K chunks [k0..k0+7] and [k0+16..k0+23],
// k0 = 8*(lane>>4), matching the CDNA5 16-bit operand VGPR striping.
__device__ inline v16bf load_frag(const __bf16* row_base, int k0) {
  FragU f;
  f.q[0] = *reinterpret_cast<const v4u*>(row_base + k0);
  f.q[1] = *reinterpret_cast<const v4u*>(row_base + k0 + 16);
  return f.v;
}

__device__ inline v8f wmma_bf16(v16bf a, v16bf b, v8f c) {
  return __builtin_amdgcn_wmma_f32_16x16x32_bf16(false, a, false, b,
                                                 (short)0, c, false, false);
}

// CDNA5 async copy: 16B per lane, global -> LDS, tracked by ASYNCcnt.
__device__ inline void async_b128(unsigned lds_off, const void* gaddr) {
  asm volatile("global_load_async_to_lds_b128 %0, %1, off"
               :: "v"(lds_off), "v"(gaddr) : "memory");
}
__device__ inline void wait_async0() {
  asm volatile("s_wait_asynccnt 0x0" ::: "memory");
}

__device__ inline float sigmoidf(float x) { return 1.0f / (1.0f + __expf(-x)); }

} // namespace

// ---- 1) fp32 -> bf16 hi/lo split (weights) ---------------------------------
__global__ void __launch_bounds__(256)
split_f32_bf16x2_kernel(const float* __restrict__ src,
                        __bf16* __restrict__ hi, __bf16* __restrict__ lo,
                        long n4) {
  long i = (long)blockIdx.x * blockDim.x + threadIdx.x;
  if (i >= n4) return;
  v4f v = reinterpret_cast<const v4f*>(src)[i];
  v4bf h, l;
#pragma unroll
  for (int j = 0; j < 4; ++j) {
    float f = v[j];
    __bf16 hb = (__bf16)f;
    h[j] = hb;
    l[j] = (__bf16)(f - (float)hb);
  }
  *reinterpret_cast<v4bf*>(hi + 4 * i) = h;
  *reinterpret_cast<v4bf*>(lo + 4 * i) = l;
}

// ---- 2) embedding gather + hi/lo split -------------------------------------
__global__ void __launch_bounds__(256)
embed_split_kernel(const int* __restrict__ tokens, const float* __restrict__ emb,
                   __bf16* __restrict__ hi, __bf16* __restrict__ lo) {
  int idx = blockIdx.x * blockDim.x + threadIdx.x;   // one thread per 4 elems
  int row = idx >> 8;                                // kH/4 = 256 chunks/row
  int c4  = (idx & 255) << 2;
  long src_off = (long)tokens[row] * kH + c4;
  v4f v = *reinterpret_cast<const v4f*>(emb + src_off);
  v4bf h, l;
#pragma unroll
  for (int j = 0; j < 4; ++j) {
    float f = v[j];
    __bf16 hb = (__bf16)f;
    h[j] = hb;
    l[j] = (__bf16)(f - (float)hb);
  }
  long dst = (long)row * kH + c4;
  *reinterpret_cast<v4bf*>(hi + dst) = h;
  *reinterpret_cast<v4bf*>(lo + dst) = l;
}

// ---- 3) gate bias: gbias[l][n] = h0[l]·Whh[l][n] + bih + bhh ---------------
__global__ void __launch_bounds__(256)
gate_bias_kernel(const float* __restrict__ Whh, const float* __restrict__ bih,
                 const float* __restrict__ bhh, const float* __restrict__ h0,
                 float* __restrict__ gbias) {
  int w    = (blockIdx.x * blockDim.x + threadIdx.x) >> 5;  // wave id 0..8191
  int lane = threadIdx.x & 31;
  int l = w >> 12;
  int n = w & (kG - 1);
  const float* wr = Whh + ((long)l * kG + n) * kH;
  const float* hv = h0 + l * kH;
  float s = 0.0f;
  for (int k = lane; k < kH; k += 32) s += hv[k] * wr[k];
#pragma unroll
  for (int off = 16; off > 0; off >>= 1) s += __shfl_xor(s, off, 32);
  if (lane == 0) gbias[l * kG + n] = s + bih[l * kG + n] + bhh[l * kG + n];
}

// ---- 4) fused LSTM layer: bf16x3 WMMA GEMM + gate epilogue -----------------
// Block: 256 threads (8 waves). Block tile: 128 rows x 32 cols (x 4 gates).
// Wave (wm = wid&3, wj = wid>>2) owns two 16-row sub-tiles x 16 cols x 4 gates
// -> 8 f32 accumulators; the full LSTM nonlinearity is applied in-register.
// B (weights, 4x shared across waves) is double-buffered in LDS and filled
// with global_load_async_to_lds_b128 overlapping the current k-step's WMMAs;
// A (rows distinct per wave) streams directly from global.
__global__ void __launch_bounds__(256)
lstm_layer_kernel(const __bf16* __restrict__ Ahi, const __bf16* __restrict__ Alo,
                  const __bf16* __restrict__ Whi, const __bf16* __restrict__ Wlo,
                  const float* __restrict__ gbias_l, const float* __restrict__ c0_l,
                  float* __restrict__ out_f32,          // non-null: final layer
                  __bf16* __restrict__ oh_hi,           // non-null: intermediate
                  __bf16* __restrict__ oh_lo) {
  // [buffer][hi/lo][128 rows * padded stride] = 2*2*128*40*2B = 40 KB
  __shared__ __bf16 ldsB[2][2][128 * kLdsRS];

  const int tid  = threadIdx.x;
  const int lane = tid & 31;
  const int wid  = tid >> 5;
  const int wm   = wid & 3;
  const int wj   = wid >> 2;
  const int half = lane >> 4;
  const int r    = lane & 15;

  const int jblk = blockIdx.y * 32;               // block col base in [0,1024)
  const int j0   = jblk + wj * 16;
  const int m0[2] = { (int)blockIdx.x * 128 + wm * 16,
                      (int)blockIdx.x * 128 + wm * 16 + 64 };

  const __bf16* arow_hi[2];
  const __bf16* arow_lo[2];
#pragma unroll
  for (int s = 0; s < 2; ++s) {
    arow_hi[s] = Ahi + (long)(m0[s] + r) * kH;
    arow_lo[s] = Alo + (long)(m0[s] + r) * kH;
  }

  // Cooperative async fill of one LDS buffer with this block's B tile
  // (4 gates x 32 cols x 32 K, hi+lo): 1024 16B chunks / 256 threads.
  auto fill = [&](int buf, int kbase) {
#pragma unroll
    for (int i = 0; i < 4; ++i) {
      int c    = i * 256 + tid;          // chunk id 0..1023
      int a    = c >> 9;                 // 0 = hi, 1 = lo (wave-uniform)
      int rem  = c & 511;
      int rowt = rem >> 2;               // tile row 0..127 (= gate*32 + col)
      int q    = rem & 3;                // 16B quarter within 64B row chunk
      long n   = (long)(rowt >> 5) * kH + jblk + (rowt & 31);  // W row
      const __bf16* src = (a ? Wlo : Whi) + n * kH + kbase + q * 8;
      unsigned loff =
          (unsigned)(unsigned long long)&ldsB[buf][a][rowt * kLdsRS + q * 8];
      async_b128(loff, src);
    }
  };

  v8f acc[2][4] = {};   // [m-subtile][gate i,f,g,o]

  fill(0, 0);
  wait_async0();
  __syncthreads();

  for (int t = 0; t < kH / kKS; ++t) {
    const int cur = t & 1;
    if (t + 1 < kH / kKS) fill(cur ^ 1, (t + 1) * kKS);

    const int k0l = 8 * half;          // within-tile K offset
    const int kg  = t * kKS + k0l;     // global K offset for A

    v16bf bh[4], bl[4];
#pragma unroll
    for (int g = 0; g < 4; ++g) {
      const __bf16* ph = &ldsB[cur][0][(g * 32 + wj * 16 + r) * kLdsRS];
      const __bf16* pl = &ldsB[cur][1][(g * 32 + wj * 16 + r) * kLdsRS];
      bh[g] = load_frag(ph, k0l);
      bl[g] = load_frag(pl, k0l);
    }
#pragma unroll
    for (int s = 0; s < 2; ++s) {
      v16bf ah = load_frag(arow_hi[s], kg);
      v16bf al = load_frag(arow_lo[s], kg);
#pragma unroll
      for (int g = 0; g < 4; ++g) {
        // bf16x3 error-compensated fp32 product: lo*hi + hi*lo + hi*hi
        acc[s][g] = wmma_bf16(al, bh[g], acc[s][g]);
        acc[s][g] = wmma_bf16(ah, bl[g], acc[s][g]);
        acc[s][g] = wmma_bf16(ah, bh[g], acc[s][g]);
      }
    }
    wait_async0();
    __syncthreads();
  }

  // ---- epilogue: per-column constants, then 8 rows per lane per subtile ----
  const int j  = j0 + r;
  const float bi  = gbias_l[0 * kH + j];
  const float bff = gbias_l[1 * kH + j];
  const float bg  = gbias_l[2 * kH + j];
  const float bo  = gbias_l[3 * kH + j];
  const float c0v = c0_l[j];

#pragma unroll
  for (int s = 0; s < 2; ++s) {
#pragma unroll
    for (int e = 0; e < 8; ++e) {
      const int m = m0[s] + e + 8 * half;   // C/D layout: VGPR e -> M = e+8*half
      float gi = sigmoidf(acc[s][0][e] + bi);
      float gf = sigmoidf(acc[s][1][e] + bff);
      float gg = tanhf   (acc[s][2][e] + bg);
      float go = sigmoidf(acc[s][3][e] + bo);
      float c  = gf * c0v + gi * gg;
      float h  = go * tanhf(c);
      long  o  = (long)m * kH + j;
      if (out_f32) {                        // final layer: outputs and h_n
        out_f32[o] = h;
        out_f32[(long)kL * kH + o] = h;
      }
      if (oh_hi) {                          // intermediate: split for next GEMM
        __bf16 hb = (__bf16)h;
        oh_hi[o] = hb;
        oh_lo[o] = (__bf16)(h - (float)hb);
      }
    }
  }
}

// ---- host launch -----------------------------------------------------------
extern "C" void kernel_launch(void* const* d_in, const int* in_sizes, int n_in,
                              void* d_out, int out_size, void* d_ws, size_t ws_size,
                              hipStream_t stream) {
  (void)in_sizes; (void)n_in; (void)out_size; (void)ws_size;

  const int*   tokens = (const int*)  d_in[0];
  const float* emb    = (const float*)d_in[1];
  const float* Wih    = (const float*)d_in[2];
  const float* Whh    = (const float*)d_in[3];
  const float* bih    = (const float*)d_in[4];
  const float* bhh    = (const float*)d_in[5];
  const float* h0     = (const float*)d_in[6];
  const float* c0     = (const float*)d_in[7];
  float*       out    = (float*)d_out;

  // workspace layout (256B aligned): gbias | w_hi | w_lo | xa_hi/lo | xb_hi/lo
  char*  ws    = (char*)d_ws;
  float* gbias = (float*)ws;
  size_t off = 32768;
  const size_t wElems = (size_t)kLayers * kG * kH;   // 8,388,608
  const size_t xElems = (size_t)kL * kH;             // 8,388,608
  __bf16* w_hi  = (__bf16*)(ws + off); off += wElems * 2;
  __bf16* w_lo  = (__bf16*)(ws + off); off += wElems * 2;
  __bf16* xa_hi = (__bf16*)(ws + off); off += xElems * 2;
  __bf16* xa_lo = (__bf16*)(ws + off); off += xElems * 2;
  __bf16* xb_hi = (__bf16*)(ws + off); off += xElems * 2;
  __bf16* xb_lo = (__bf16*)(ws + off); off += xElems * 2;

  // 1) split Wih (both layers) into bf16 hi/lo
  {
    long n4 = (long)(wElems / 4);
    split_f32_bf16x2_kernel<<<(unsigned)((n4 + 255) / 256), 256, 0, stream>>>(
        Wih, w_hi, w_lo, n4);
  }
  // 2) embedding gather + split
  {
    int nthreads = kL * (kH / 4);
    embed_split_kernel<<<nthreads / 256, 256, 0, stream>>>(tokens, emb, xa_hi, xa_lo);
  }
  // 3) gate bias (h0·Whhᵀ + bih + bhh) for both layers: 8192 waves
  gate_bias_kernel<<<(kLayers * kG * 32) / 256, 256, 0, stream>>>(
      Whh, bih, bhh, h0, gbias);

  // 4) layer 0: emits split bf16 h for layer 1
  dim3 grid(kL / 128, kH / 32);   // 64 x 32 blocks, 256 threads
  lstm_layer_kernel<<<grid, 256, 0, stream>>>(
      xa_hi, xa_lo, w_hi, w_lo, gbias, c0,
      /*out_f32=*/nullptr, xb_hi, xb_lo);

  // 5) layer 1: writes both output halves (outputs, h_n)
  lstm_layer_kernel<<<grid, 256, 0, stream>>>(
      xb_hi, xb_lo, w_hi + (size_t)kG * kH, w_lo + (size_t)kG * kH,
      gbias + kG, c0 + kH,
      out, /*oh_hi=*/nullptr, /*oh_lo=*/nullptr);
}
